// MPMM_41704132444279
// MI455X (gfx1250) — matrix operand order repaired and checked by reading, not compiled, not run on hardware
//
#include <hip/hip_runtime.h>
#include <math.h>

// ---------------- model dims ----------------
#define BB   64
#define LLn  256
#define EEn  300
#define HHn  200
#define G4   800
#define PPn  10
#define BLr  (BB*LLn)   // 16384

// ---------------- types ----------------
typedef __attribute__((ext_vector_type(16))) __bf16          v16bf;
typedef __attribute__((ext_vector_type(16))) unsigned short  v16us;
typedef __attribute__((ext_vector_type(8)))  float           v8f;

struct f4 { float x, y, z, w; };   // 4-aligned quad load (backend may widen to b128)

__device__ __forceinline__ unsigned short f2bf(float f){
  unsigned u = __float_as_uint(f);
  u += 0x7FFFu + ((u >> 16) & 1u);
  return (unsigned short)(u >> 16);
}
__device__ __forceinline__ float geluf(float x){
  return 0.5f * x * (1.f + tanhf(x * 0.7978845608f * (1.f + 0.044715f * x * x)));
}
__device__ __forceinline__ float sigm(float x){ return 1.f / (1.f + __expf(-x)); }

__device__ __forceinline__ void atomicMaxF(float* addr, float val){
  unsigned* u = (unsigned*)addr;
  unsigned old = *u;
  while (__uint_as_float(old) < val){
    unsigned prev = atomicCAS(u, old, __float_as_uint(val));
    if (prev == old) break;
    old = prev;
  }
}

// ---------------- WMMA fragment helpers (CDNA5 ISA 7.12.2 layouts) ----------------
__device__ __forceinline__ v16bf load_afrag(const unsigned short* Ar, int lda, int lane){
  int ml = lane & 15, hi = lane >> 4;
  const unsigned short* ap = Ar + ml * lda;
  v16us au;
#pragma unroll
  for (int e = 0; e < 8; ++e){ au[e] = ap[hi*8 + e]; au[e+8] = ap[16 + hi*8 + e]; }
  return __builtin_bit_cast(v16bf, au);
}
__device__ __forceinline__ v16bf load_bfrag(const unsigned short* Br, int ldb, int lane){
  int ml = lane & 15, hi = lane >> 4;
  const unsigned short* bp = Br + ml * ldb;
  v16us bu;
#pragma unroll
  for (int e = 0; e < 8; ++e){ bu[e] = bp[hi*8 + e]; bu[e+8] = bp[16 + hi*8 + e]; }
  return __builtin_bit_cast(v16bf, bu);
}
__device__ __forceinline__ v8f wmma32(v16bf a, v16bf b, v8f c){
  return __builtin_amdgcn_wmma_f32_16x16x32_bf16(false, a, false, b, (short)0, c, false, false);
}

// ---------------- f32 -> bf16 LDS tile staging helpers ----------------
// Row-contiguous quad: loads src[gr*lda + gc .. +3], stores 4 bf16 contiguously.
__device__ __forceinline__ void stage4_row(const float* src, int lda, int gr, int R,
                                           int gc, int C, unsigned short* dst){
  unsigned short h0, h1, h2, h3;
  if (gr < R && gc + 3 < C){
    f4 v = *(const f4*)(src + (long)gr * lda + gc);
    h0 = f2bf(v.x); h1 = f2bf(v.y); h2 = f2bf(v.z); h3 = f2bf(v.w);
  } else {
    float x0 = (gr < R && gc + 0 < C) ? src[(long)gr*lda + gc + 0] : 0.f;
    float x1 = (gr < R && gc + 1 < C) ? src[(long)gr*lda + gc + 1] : 0.f;
    float x2 = (gr < R && gc + 2 < C) ? src[(long)gr*lda + gc + 2] : 0.f;
    float x3 = (gr < R && gc + 3 < C) ? src[(long)gr*lda + gc + 3] : 0.f;
    h0 = f2bf(x0); h1 = f2bf(x1); h2 = f2bf(x2); h3 = f2bf(x3);
  }
  uint2 pk;
  pk.x = (unsigned)h0 | ((unsigned)h1 << 16);
  pk.y = (unsigned)h2 | ((unsigned)h3 << 16);
  *(uint2*)dst = pk;   // 8B-aligned LDS store
}
// Transposed quad: loads src[gk*lda + gm .. +3] (4 consecutive logical rows),
// scatters into dst[j*dstStride].
__device__ __forceinline__ void stage4_col(const float* src, int lda, int gk, int K,
                                           int gm, int M, unsigned short* dst, int dstStride){
  unsigned short h[4];
  if (gk < K && gm + 3 < M){
    f4 v = *(const f4*)(src + (long)gk * lda + gm);
    h[0] = f2bf(v.x); h[1] = f2bf(v.y); h[2] = f2bf(v.z); h[3] = f2bf(v.w);
  } else {
#pragma unroll
    for (int j = 0; j < 4; ++j){
      float x = (gk < K && gm + j < M) ? src[(long)gk*lda + gm + j] : 0.f;
      h[j] = f2bf(x);
    }
  }
#pragma unroll
  for (int j = 0; j < 4; ++j) dst[j * dstStride] = h[j];
}

// ---------------- generic WMMA GEMM: C = act(alpha*A@B^T + bias) ----------------
// AT==0: A[m*lda+k], AT==1: A[k*lda+m].  BT==0: B[n*ldb+k], BT==1: B[k*ldb+n].
struct GemmP {
  const float* A; const float* B; const float* bias; float* C;
  long sA, sB, sC;
  int M, N, K, lda, ldb, ldc;
  int act;
  float alpha;
};

template<int AT, int BT>
__global__ __launch_bounds__(256) void k_gemm_t(GemmP p){
  __shared__ unsigned short As[64*32];
  __shared__ unsigned short Bs[64*32];
  int bz = blockIdx.z;
  const float* A  = p.A + (long)bz * p.sA;
  const float* Bm = p.B + (long)bz * p.sB;
  float*       C  = p.C + (long)bz * p.sC;
  int M0 = blockIdx.x * 64, N0 = blockIdx.y * 64;
  int tid = threadIdx.x;
  int w = tid >> 5, lane = tid & 31;
  int wm = w & 3, wn = w >> 2;           // 4 M-tiles x 2 N-strips (each strip 32 wide)
  v8f acc[2];
#pragma unroll
  for (int t = 0; t < 2; ++t)
#pragma unroll
    for (int q = 0; q < 8; ++q) acc[t][q] = 0.f;

  for (int k0 = 0; k0 < p.K; k0 += 32){
    // ---- stage A: 64 rows x 32 k ----
    if (AT == 0){
#pragma unroll
      for (int i = 0; i < 2; ++i){
        int idx = tid + i*256;
        int r = idx >> 3, c4 = idx & 7;
        int gm = M0 + r, gk = k0 + c4*4;
        if (i == 0 && gm < p.M && gk + 32 < p.K)
          __builtin_prefetch(A + (long)gm*p.lda + gk + 32, 0, 1);
        stage4_row(A, p.lda, gm, p.M, gk, p.K, &As[r*32 + c4*4]);
      }
    } else {
#pragma unroll
      for (int i = 0; i < 2; ++i){
        int idx = tid + i*256;
        int c = idx >> 4, m4 = idx & 15;
        stage4_col(A, p.lda, k0 + c, p.K, M0 + m4*4, p.M, &As[(m4*4)*32 + c], 32);
      }
    }
    // ---- stage B: 64 n-rows x 32 k ----
    if (BT == 0){
#pragma unroll
      for (int i = 0; i < 2; ++i){
        int idx = tid + i*256;
        int r = idx >> 3, c4 = idx & 7;
        int gn = N0 + r, gk = k0 + c4*4;
        if (i == 0 && gn < p.N && gk + 32 < p.K)
          __builtin_prefetch(Bm + (long)gn*p.ldb + gk + 32, 0, 1);
        stage4_row(Bm, p.ldb, gn, p.N, gk, p.K, &Bs[r*32 + c4*4]);
      }
    } else {
#pragma unroll
      for (int i = 0; i < 2; ++i){
        int idx = tid + i*256;
        int c = idx >> 4, n4 = idx & 15;
        stage4_col(Bm, p.ldb, k0 + c, p.K, N0 + n4*4, p.N, &Bs[(n4*4)*32 + c], 32);
      }
    }
    __syncthreads();
    v16bf af  = load_afrag(&As[(16*wm)*32], 32, lane);
    v16bf bf0 = load_bfrag(&Bs[(32*wn)*32], 32, lane);
    v16bf bf1 = load_bfrag(&Bs[(32*wn + 16)*32], 32, lane);
    acc[0] = wmma32(af, bf0, acc[0]);
    acc[1] = wmma32(af, bf1, acc[1]);
    __syncthreads();
  }
  int ml = lane & 15, hi = lane >> 4;
#pragma unroll
  for (int t = 0; t < 2; ++t){
    int n = N0 + 32*wn + 16*t + ml;
#pragma unroll
    for (int r = 0; r < 8; ++r){
      int m = M0 + 16*wm + r + 8*hi;
      if (m < p.M && n < p.N){
        float v = acc[t][r] * p.alpha;
        if (p.bias) v += p.bias[n];
        if (p.act)  v = geluf(v);
        C[(long)m * p.ldc + n] = v;
      }
    }
  }
}

// ---------------- persistent bi-LSTM direction kernel ----------------
struct LstmInst {
  const float* pre;    // [B, L, 800]
  const float* Whh;    // [800, 200]
  const float* bhh;    // [800]
  float* hs;           // [B, L, 200] or null
  float* hlast;        // row-strided or null
  float* gates;        // [64*800] scratch
  float* c;            // [64*200] scratch
  int reverse;
  int hlStride;
};
struct LstmArgs { LstmInst inst[4]; int n; int L; };

__global__ __launch_bounds__(512) void k_lstm(LstmArgs A){
  LstmInst I = A.inst[blockIdx.x];
  __shared__ unsigned short hbf[64*224];   // h (bf16), K padded 200->224
  __shared__ unsigned short wbf[64*224];   // Whh chunk (64 gate rows)
  const int Kp = 224;
  int tid = threadIdx.x;
  for (int i = tid; i < 64*224; i += 512) hbf[i] = 0;
  for (int i = tid; i < 64*200; i += 512) I.c[i] = 0.f;
  __threadfence();
  __syncthreads();
  int w = tid >> 5, lane = tid & 31;
  int wm = w & 3, wn = w >> 2;      // 4 M-tiles x 4 N-tiles = 64x64
  int ml = lane & 15, hi = lane >> 4;

  for (int ti = 0; ti < A.L; ++ti){
    int t = I.reverse ? (A.L - 1 - ti) : ti;
    // phase 1: gates = h @ Whh^T + pre + bhh
    for (int n0 = 0; n0 < G4; n0 += 64){
      __syncthreads();
#pragma unroll
      for (int i = 0; i < 7; ++i){            // 64 rows x 56 quads = 3584 / 512
        int idx = tid + i*512;
        int r = idx / 56, c4 = idx % 56;
        stage4_row(I.Whh, HHn, n0 + r, G4, c4*4, HHn, &wbf[r*Kp + c4*4]);
      }
      __syncthreads();
      v8f acc;
#pragma unroll
      for (int q = 0; q < 8; ++q) acc[q] = 0.f;
#pragma unroll
      for (int kc = 0; kc < 7; ++kc){
        v16bf af = load_afrag(&hbf[(16*wm)*Kp + kc*32], Kp, lane);
        v16bf bf = load_bfrag(&wbf[(16*wn)*Kp + kc*32], Kp, lane);
        acc = wmma32(af, bf, acc);
      }
#pragma unroll
      for (int r = 0; r < 8; ++r){
        int m = 16*wm + r + 8*hi;        // batch index
        int n = n0 + 16*wn + ml;         // gate index
        if (n < G4){
          float v = acc[r] + I.pre[((long)m * A.L + t) * G4 + n] + I.bhh[n];
          I.gates[m * G4 + n] = v;
        }
      }
    }
    __threadfence();
    __syncthreads();
    // phase 2: gate nonlinearity, update c, h
    for (int i = tid; i < BB*HHn; i += 512){
      int bb = i / HHn, u = i % HHn;
      const float* g = I.gates + bb * G4;
      float gi = g[u], gf = g[200+u], gg = g[400+u], go = g[600+u];
      float cc = sigm(gf) * I.c[i] + sigm(gi) * tanhf(gg);
      float hh = sigm(go) * tanhf(cc);
      I.c[i] = cc;
      hbf[bb*224 + u] = f2bf(hh);
      if (I.hs)    I.hs[((long)bb * A.L + t) * HHn + u] = hh;
      if (I.hlast) I.hlast[(long)bb * I.hlStride + u] = hh;
    }
    __threadfence();
    __syncthreads();
  }
}

// ---------------- utility kernels ----------------
__global__ void k_embed(const int* words, const float* emb, float* out, float* mask){
  long row = blockIdx.x;
  int id = words[row];
  const float* src = emb + (long)id * EEn;
  for (int c = threadIdx.x; c < EEn; c += blockDim.x) out[row*EEn + c] = src[c];
  if (threadIdx.x == 0) mask[row] = (src[0] != 0.f) ? 1.f : 0.f;
}

__global__ __launch_bounds__(256) void k_softmax(const float* in, float* out, int innerCnt,
                                                 long sHi, long sLo, long sAx, int len){
  long o = blockIdx.x;
  long base = (o / innerCnt) * sHi + (o % innerCnt) * sLo;
  __shared__ float red[256];
  float m = -1e30f;
  for (int i = threadIdx.x; i < len; i += 256) m = fmaxf(m, in[base + i*sAx]);
  red[threadIdx.x] = m; __syncthreads();
  for (int st = 128; st > 0; st >>= 1){
    if (threadIdx.x < st) red[threadIdx.x] = fmaxf(red[threadIdx.x], red[threadIdx.x + st]);
    __syncthreads();
  }
  m = red[0]; __syncthreads();
  float ssum = 0.f;
  for (int i = threadIdx.x; i < len; i += 256){
    float e = __expf(in[base + i*sAx] - m);
    out[base + i*sAx] = e; ssum += e;
  }
  red[threadIdx.x] = ssum; __syncthreads();
  for (int st = 128; st > 0; st >>= 1){
    if (threadIdx.x < st) red[threadIdx.x] += red[threadIdx.x + st];
    __syncthreads();
  }
  float inv = 1.f / red[0];
  for (int i = threadIdx.x; i < len; i += 256) out[base + i*sAx] *= inv;
}

// mode: 0=sum, 1=max, 2=l2norm
__global__ __launch_bounds__(256) void k_reduce(const float* in, float* out, int innerCnt,
                                                long sHi, long sLo, long sAx, int len, int mode){
  long o = blockIdx.x;
  long base = (o / innerCnt) * sHi + (o % innerCnt) * sLo;
  __shared__ float red[256];
  float acc = (mode == 1) ? -1e30f : 0.f;
  for (int i = threadIdx.x; i < len; i += 256){
    float v = in[base + i*sAx];
    if (mode == 0) acc += v;
    else if (mode == 1) acc = fmaxf(acc, v);
    else acc += v*v;
  }
  red[threadIdx.x] = acc; __syncthreads();
  for (int st = 128; st > 0; st >>= 1){
    if (threadIdx.x < st){
      float a = red[threadIdx.x], b = red[threadIdx.x + st];
      red[threadIdx.x] = (mode == 1) ? fmaxf(a, b) : (a + b);
    }
    __syncthreads();
  }
  if (threadIdx.x == 0) out[o] = (mode == 2) ? sqrtf(red[0]) : red[0];
}

__global__ void k_copy(const float* src, int lds, float* dst, int ldd, long rows, int w){
  long i = blockIdx.x * (long)blockDim.x + threadIdx.x;
  if (i >= rows * (long)w) return;
  long r = i / w; int c = (int)(i % w);
  dst[r * ldd + c] = src[r * lds + c];
}

__global__ void k_sub(const float* a, const float* b, float* o, long n){
  long i = blockIdx.x * (long)blockDim.x + threadIdx.x;
  if (i < n) o[i] = a[i] - b[i];
}

__global__ void k_divrow(float* x, const float* d, long rows, int w){
  long i = blockIdx.x * (long)blockDim.x + threadIdx.x;
  if (i >= rows * (long)w) return;
  x[i] /= d[i / w];
}

__global__ void k_fill(float* dst, long rows, int ld, int w, float val){
  long i = blockIdx.x * (long)blockDim.x + threadIdx.x;
  if (i >= rows * (long)w) return;
  dst[(i / w) * ld + (i % w)] = val;
}

__global__ void k_masktemp(const float* raw, float* out, const float* ma, const float* mb,
                           const float* temp, long tot){
  long i = blockIdx.x * (long)blockDim.x + threadIdx.x;
  if (i >= tot) return;
  int m = (int)(i & 255); long r = i >> 8; int l = (int)(r & 255); int b = (int)(r >> 8);
  float pair = ma[b*LLn + l] * mb[b*LLn + m];
  out[i] = (pair > 0.f) ? raw[i] * temp[0] : -1e7f;
}

__global__ void k_alphadiv(float* attn, const float* na, const float* nb, long tot){
  long i = blockIdx.x * (long)blockDim.x + threadIdx.x;
  if (i >= tot) return;
  int m = (int)(i & 255); long r = i >> 8; int l = (int)(r & 255); int b = (int)(r >> 8);
  attn[i] = attn[i] / (na[b*LLn + l] * nb[b*LLn + m] + 1e-8f);
}

// weighted cosine: out[b,l,p] = cos(x[b,l,:]*w_p, y*w_p); y strided for broadcasts
__global__ void k_wcos(const float* x, long xsb, const float* y, long ysb, long ysl, long ysh,
                       const float* w, float* out, long osb, int old_, long tot){
  long i = blockIdx.x * (long)blockDim.x + threadIdx.x;
  if (i >= tot) return;
  int p = (int)(i % PPn); long bl = i / PPn; int l = (int)(bl % LLn); int b = (int)(bl / LLn);
  const float* xr = x + (long)b*xsb + (long)l*HHn;
  const float* yr = y + (long)b*ysb + (long)l*ysl;
  const float* wr = w + (long)p*HHn;
  float dot = 0.f, nx = 0.f, ny = 0.f;
  for (int h = 0; h < HHn; ++h){
    float wv = wr[h]*wr[h];
    float xv = xr[h];
    float yv = yr[(long)h*ysh];
    dot += xv*yv*wv; nx += xv*xv*wv; ny += yv*yv*wv;
  }
  float den = sqrtf(nx) * sqrtf(ny);
  out[(long)b*osb + (long)l*old_ + p] = dot / fmaxf(den, 1e-8f);
}

// weighted norms: out[(b*P+p)*L + l] = sqrt(sum_h x^2 * w_p^2)
__global__ void k_wn(const float* x, const float* w2, float* out, long tot){
  long i = blockIdx.x * (long)blockDim.x + threadIdx.x;
  if (i >= tot) return;
  int l = (int)(i % LLn); long t = i / LLn; int p = (int)(t % PPn); int b = (int)(t / PPn);
  const float* xr = x + ((long)b*LLn + l) * HHn;
  const float* wr = w2 + (long)p*HHn;
  float s = 0.f;
  for (int h = 0; h < HHn; ++h){ float wv = wr[h]*wr[h]; s += xr[h]*xr[h]*wv; }
  out[i] = sqrtf(s);
}

// max-pool matching: mm_a[b,l,p] = max_m wcos, mm_b[b,m,p] = max_l wcos (atomic).
// B-tiles staged with CDNA5 async global->LDS copies (ASYNCcnt).
__global__ __launch_bounds__(256) void k_mpm(const float* a, const float* bM, const float* w2,
                                             const float* wna, const float* wnb,
                                             float* mma, float* mmb){
  __shared__ __align__(16) float asub[16*HHn];
  __shared__ __align__(16) float bsub[16*HHn];
  __shared__ float red[256];
  int b = blockIdx.z, p = blockIdx.y, lt = blockIdx.x;
  const float* wr = w2 + (long)p*HHn;
  for (int i = threadIdx.x; i < 16*HHn; i += 256){
    int r = i / HHn, h = i % HHn;
    float wv = wr[h]*wr[h];
    asub[i] = a[((long)b*LLn + lt*16 + r) * HHn + h] * wv;
  }
  int lr = threadIdx.x >> 4, mr = threadIdx.x & 15;
  int l = lt*16 + lr;
  float n1 = wna[((long)b*PPn + p)*LLn + l];
  float mymax = -1e30f;
  for (int m0 = 0; m0 < LLn; m0 += 16){
    __syncthreads();
    // async copy of the 16x200 f32 b-tile into LDS (800 x b128)
    for (int i = threadIdx.x; i < 16*HHn/4; i += 256){
      int r = i / 50, c4 = i % 50;
      const float* gp = &bM[((long)b*LLn + m0 + r) * HHn + c4*4];
      unsigned lds = (unsigned)(unsigned long long)(&bsub[r*HHn + c4*4]);
      asm volatile("global_load_async_to_lds_b128 %0, %1, off"
                   :: "v"(lds), "v"(gp) : "memory");
    }
    asm volatile("s_wait_asynccnt 0" ::: "memory");
    __syncthreads();
    float dot = 0.f;
    const float* ar = asub + lr*HHn;
    const float* br = bsub + mr*HHn;
    for (int h = 0; h < HHn; ++h) dot += ar[h]*br[h];
    int m = m0 + mr;
    float n2 = wnb[((long)b*PPn + p)*LLn + m];
    float cv = dot / (n1*n2 + 1e-8f);
    mymax = fmaxf(mymax, cv);
    red[threadIdx.x] = cv;
    __syncthreads();
    if (lr == 0){
      float mx = -1e30f;
      for (int j = 0; j < 16; ++j) mx = fmaxf(mx, red[j*16 + mr]);
      atomicMaxF(&mmb[((long)b*LLn + m)*40 + p], mx);
    }
  }
  __syncthreads();
  red[threadIdx.x] = mymax;
  __syncthreads();
  if (mr == 0){
    float mx = -1e30f;
    for (int j = 0; j < 16; ++j) mx = fmaxf(mx, red[lr*16 + j]);
    mma[((long)b*LLn + l)*40 + p] = mx;
  }
}

// ---------------- host orchestration ----------------
static inline int gblk(long n){ return (int)((n + 255) / 256); }

static void launch_gemm(hipStream_t st, const float* A, const float* B, const float* bias, float* C,
                        int M, int N, int K, int lda, int ldb, int ldc, int aT, int bT, int act,
                        float alpha, int nb, long sA, long sB, long sC){
  GemmP p; p.A=A; p.B=B; p.bias=bias; p.C=C; p.sA=sA; p.sB=sB; p.sC=sC;
  p.M=M; p.N=N; p.K=K; p.lda=lda; p.ldb=ldb; p.ldc=ldc; p.act=act; p.alpha=alpha;
  dim3 grid((M+63)/64, (N+63)/64, nb);
  if      (aT == 0 && bT == 0) k_gemm_t<0,0><<<grid, 256, 0, st>>>(p);
  else if (aT == 0 && bT == 1) k_gemm_t<0,1><<<grid, 256, 0, st>>>(p);
  else if (aT == 1 && bT == 1) k_gemm_t<1,1><<<grid, 256, 0, st>>>(p);
  else                         k_gemm_t<1,0><<<grid, 256, 0, st>>>(p);
}

static void copy2d(hipStream_t st, const float* src, int lds, float* dst, int ldd, long rows, int w){
  k_copy<<<gblk(rows*(long)w), 256, 0, st>>>(src, lds, dst, ldd, rows, w);
}

static void run_lstm2(hipStream_t st, const float* preF, const float* preB,
                      const float* WhhF, const float* bhhF, const float* WhhB, const float* bhhB,
                      float* hsF, float* hsB, float* hlF, float* hlB, int hlStride,
                      float* gates, float* cbuf){
  LstmArgs a{};
  a.n = 2; a.L = LLn;
  a.inst[0] = {preF, WhhF, bhhF, hsF, hlF, gates,            cbuf,            0, hlStride};
  a.inst[1] = {preB, WhhB, bhhB, hsB, hlB, gates + 64*G4,    cbuf + 64*HHn,   1, hlStride};
  k_lstm<<<2, 512, 0, st>>>(a);
}

struct WS {
  float *embA,*embB,*maskA,*maskB,*q,*kk,*vv,*sattn,*sv,*glob,*preF,*preB,*hsF,*hsB,*catEnc,
        *aEnc,*bEnc,*gates,*cbuf,*attnRaw,*attnM,*S1,*S2,*fa,*fb,*na,*nb,*as1,*as2,*am1,*am2,
        *s1m,*s2m,*wnA,*wnB,*mA,*mB,*cat,*i1,*i2,*xmi,*z,*ph;
};

static float* bump(char*& p, size_t n){
  float* r = (float*)p;
  size_t bytes = (n * sizeof(float) + 255) & ~(size_t)255;
  p += bytes;
  return r;
}

static void enc_arnn(hipStream_t st, void* const* d_in, WS& w, const float* emb, float* xEnc){
  #define FIN(i) ((const float*)d_in[i])
  launch_gemm(st, emb, FIN(3), FIN(4), w.q,  BLr, 50, EEn, EEn, EEn, 50, 0,0,0, 1.f, 1,0,0,0);
  launch_gemm(st, emb, FIN(5), FIN(6), w.kk, BLr, 50, EEn, EEn, EEn, 50, 0,0,0, 1.f, 1,0,0,0);
  launch_gemm(st, emb, FIN(7), FIN(8), w.vv, BLr, 50, EEn, EEn, EEn, 50, 0,0,0, 1.f, 1,0,0,0);
  // s = q@k^T / sqrt(H=200), batched; row-constant mask term is softmax-invariant
  launch_gemm(st, w.q, w.kk, nullptr, w.sattn, 256,256,50, 50,50,256, 0,0,0,
              0.07071067811865475f, 64, 256*50, 256*50, 65536);
  k_softmax<<<BLr, 256, 0, st>>>(w.sattn, w.sattn, 1, 256, 0, 1, 256);
  launch_gemm(st, w.sattn, w.vv, nullptr, w.sv, 256,50,256, 256,50,50, 0,1,0, 1.f, 64, 65536, 256*50, 256*50);
  launch_gemm(st, w.sv, FIN(9), FIN(10), w.glob, BLr,50,50, 50,50,50, 0,0,1, 1.f, 1,0,0,0);
  // bilstm input projections + recurrence
  launch_gemm(st, emb, FIN(11), FIN(13), w.preF, BLr, G4, EEn, EEn, EEn, G4, 0,0,0, 1.f, 1,0,0,0);
  launch_gemm(st, emb, FIN(15), FIN(17), w.preB, BLr, G4, EEn, EEn, EEn, G4, 0,0,0, 1.f, 1,0,0,0);
  run_lstm2(st, w.preF, w.preB, FIN(12), FIN(14), FIN(16), FIN(18),
            w.hsF, w.hsB, nullptr, nullptr, 0, w.gates, w.cbuf);
  // concat [hs_f | hs_b | glob] -> Wo gelu
  copy2d(st, w.hsF, 200, w.catEnc + 0,   450, BLr, 200);
  copy2d(st, w.hsB, 200, w.catEnc + 200, 450, BLr, 200);
  copy2d(st, w.glob, 50, w.catEnc + 400, 450, BLr, 50);
  launch_gemm(st, w.catEnc, FIN(19), FIN(20), xEnc, BLr, 200, 450, 450, 450, 200, 0,0,1, 1.f, 1,0,0,0);
  #undef FIN
}

static void fusion_side(hipStream_t st, void* const* d_in, WS& w,
                        const float* x, const float* ii, const float* m, float* zbase){
  #define FIN(i) ((const float*)d_in[i])
  long n200 = (long)BLr * 200;
  copy2d(st, x,  200, w.cat + 0,   400, BLr, 200);
  copy2d(st, ii, 200, w.cat + 200, 400, BLr, 200);
  launch_gemm(st, w.cat, FIN(26), FIN(27), w.i1, BLr, 200, 400, 400, 400, 200, 0,0,1, 1.f, 1,0,0,0);
  k_sub<<<gblk(n200), 256, 0, st>>>(x, ii, w.xmi, n200);
  copy2d(st, x,     200, w.cat + 0,   400, BLr, 200);
  copy2d(st, w.xmi, 200, w.cat + 200, 400, BLr, 200);
  launch_gemm(st, w.cat, FIN(28), FIN(29), w.i2, BLr, 200, 400, 400, 400, 200, 0,0,1, 1.f, 1,0,0,0);
  copy2d(st, w.i1, 200, w.cat + 0,   400, BLr, 200);
  copy2d(st, w.i2, 200, w.cat + 200, 400, BLr, 200);
  // lstm_i: only h_last needed
  launch_gemm(st, w.cat, FIN(30), FIN(32), w.preF, BLr, G4, 400, 400, 400, G4, 0,0,0, 1.f, 1,0,0,0);
  launch_gemm(st, w.cat, FIN(34), FIN(36), w.preB, BLr, G4, 400, 400, 400, G4, 0,0,0, 1.f, 1,0,0,0);
  run_lstm2(st, w.preF, w.preB, FIN(31), FIN(33), FIN(35), FIN(37),
            nullptr, nullptr, zbase + 0, zbase + 400, 1600, w.gates, w.cbuf);
  // lstm_m on matching vector [B,L,40]
  launch_gemm(st, m, FIN(38), FIN(40), w.preF, BLr, G4, 40, 40, 40, G4, 0,0,0, 1.f, 1,0,0,0);
  launch_gemm(st, m, FIN(42), FIN(44), w.preB, BLr, G4, 40, 40, 40, G4, 0,0,0, 1.f, 1,0,0,0);
  run_lstm2(st, w.preF, w.preB, FIN(39), FIN(41), FIN(43), FIN(45),
            nullptr, nullptr, zbase + 200, zbase + 600, 1600, w.gates, w.cbuf);
  #undef FIN
}

extern "C" void kernel_launch(void* const* d_in, const int* in_sizes, int n_in,
                              void* d_out, int out_size, void* d_ws, size_t ws_size,
                              hipStream_t stream){
  (void)in_sizes; (void)n_in; (void)out_size; (void)ws_size;
  #define FIN(i) ((const float*)d_in[i])
  char* pw = (char*)d_ws;
  WS w;
  w.embA   = bump(pw, (size_t)BLr*EEn);  w.embB  = bump(pw, (size_t)BLr*EEn);
  w.maskA  = bump(pw, BLr);              w.maskB = bump(pw, BLr);
  w.q      = bump(pw, (size_t)BLr*50);   w.kk    = bump(pw, (size_t)BLr*50);
  w.vv     = bump(pw, (size_t)BLr*50);   w.sv    = bump(pw, (size_t)BLr*50);
  w.glob   = bump(pw, (size_t)BLr*50);
  w.sattn  = bump(pw, (size_t)BB*LLn*LLn);
  w.preF   = bump(pw, (size_t)BLr*G4);   w.preB  = bump(pw, (size_t)BLr*G4);
  w.hsF    = bump(pw, (size_t)BLr*HHn);  w.hsB   = bump(pw, (size_t)BLr*HHn);
  w.catEnc = bump(pw, (size_t)BLr*450);
  w.aEnc   = bump(pw, (size_t)BLr*HHn);  w.bEnc  = bump(pw, (size_t)BLr*HHn);
  w.gates  = bump(pw, (size_t)2*64*G4);  w.cbuf  = bump(pw, (size_t)2*64*HHn);
  w.attnRaw= bump(pw, (size_t)BB*LLn*LLn);
  w.attnM  = bump(pw, (size_t)BB*LLn*LLn);
  w.S1     = bump(pw, (size_t)BB*LLn*LLn);
  w.S2     = bump(pw, (size_t)BB*LLn*LLn);
  w.fa     = bump(pw, (size_t)BLr*HHn);  w.fb    = bump(pw, (size_t)BLr*HHn);
  w.na     = bump(pw, BLr);  w.nb  = bump(pw, BLr);
  w.as1    = bump(pw, BLr);  w.as2 = bump(pw, BLr);
  w.am1    = bump(pw, BLr);  w.am2 = bump(pw, BLr);
  w.s1m    = bump(pw, (size_t)BLr*HHn);  w.s2m   = bump(pw, (size_t)BLr*HHn);
  w.wnA    = bump(pw, (size_t)BB*PPn*LLn); w.wnB = bump(pw, (size_t)BB*PPn*LLn);
  w.mA     = bump(pw, (size_t)BLr*40);   w.mB    = bump(pw, (size_t)BLr*40);
  w.cat    = bump(pw, (size_t)BLr*400);
  w.i1     = bump(pw, (size_t)BLr*HHn);  w.i2    = bump(pw, (size_t)BLr*HHn);
  w.xmi    = bump(pw, (size_t)BLr*HHn);
  w.z      = bump(pw, (size_t)64*1600);  w.ph    = bump(pw, (size_t)64*400);

  // ---- embeddings + masks ----
  k_embed<<<BLr, 128, 0, stream>>>((const int*)d_in[0], FIN(2), w.embA, w.maskA);
  k_embed<<<BLr, 128, 0, stream>>>((const int*)d_in[1], FIN(2), w.embB, w.maskB);

  // ---- encoder (arnn) for both sides ----
  enc_arnn(stream, d_in, w, w.embA, w.aEnc);
  enc_arnn(stream, d_in, w, w.embB, w.bEnc);

  // ---- multi-perspective matching ----
  long totLL = (long)BB * LLn * LLn;
  launch_gemm(stream, w.aEnc, w.bEnc, nullptr, w.attnRaw, 256,256,200, 200,200,256,
              0,0,0, 1.f, 64, 51200, 51200, 65536);
  k_masktemp<<<gblk(totLL), 256, 0, stream>>>(w.attnRaw, w.attnM, w.maskA, w.maskB, FIN(21), totLL);
  k_softmax<<<BLr, 256, 0, stream>>>(w.attnM, w.S1, 256, 65536, 1, 256, 256);  // over dim 1 (l)
  k_softmax<<<BLr, 256, 0, stream>>>(w.attnM, w.S2, 1,   256,   0, 1,   256);  // over dim 2 (m)
  launch_gemm(stream, w.S1, w.aEnc, nullptr, w.fb, 256,200,256, 256,200,200,
              1,1,0, 1.f, 64, 65536, 51200, 51200);
  launch_gemm(stream, w.S2, w.bEnc, nullptr, w.fa, 256,200,256, 256,200,200,
              0,1,0, 1.f, 64, 65536, 51200, 51200);
  // alpha = raw / (|a||b| + eps)
  k_reduce<<<BLr, 256, 0, stream>>>(w.aEnc, w.na, 1, 200, 0, 1, 200, 2);
  k_reduce<<<BLr, 256, 0, stream>>>(w.bEnc, w.nb, 1, 200, 0, 1, 200, 2);
  k_alphadiv<<<gblk(totLL), 256, 0, stream>>>(w.attnRaw, w.na, w.nb, totLL);
  k_reduce<<<BLr, 256, 0, stream>>>(w.attnRaw, w.as2, 1,   256,   0, 1,   256, 0);
  k_reduce<<<BLr, 256, 0, stream>>>(w.attnRaw, w.as1, 256, 65536, 1, 256, 256, 0);
  launch_gemm(stream, w.attnRaw, w.bEnc, nullptr, w.s2m, 256,200,256, 256,200,200,
              0,1,0, 1.f, 64, 65536, 51200, 51200);
  k_divrow<<<gblk((long)BLr*200), 256, 0, stream>>>(w.s2m, w.as2, BLr, 200);
  launch_gemm(stream, w.attnRaw, w.aEnc, nullptr, w.s1m, 256,200,256, 256,200,200,
              1,1,0, 1.f, 64, 65536, 51200, 51200);
  k_divrow<<<gblk((long)BLr*200), 256, 0, stream>>>(w.s1m, w.as1, BLr, 200);
  k_reduce<<<BLr, 256, 0, stream>>>(w.attnRaw, w.am2, 1,   256,   0, 1,   256, 1);
  k_reduce<<<BLr, 256, 0, stream>>>(w.attnRaw, w.am1, 256, 65536, 1, 256, 256, 1);
  // weighted cosine matchings -> m_a/m_b columns [fm|mm|am|mam]
  long totW = (long)BB * LLn * PPn;
  k_wcos<<<gblk(totW), 256, 0, stream>>>(w.aEnc, 51200, w.bEnc + 255*200, 51200, 0, 1,
                                         FIN(22), w.mA + 0,  LLn*40, 40, totW);
  k_wcos<<<gblk(totW), 256, 0, stream>>>(w.aEnc, 51200, w.s2m, 51200, 200, 1,
                                         FIN(24), w.mA + 20, LLn*40, 40, totW);
  k_wcos<<<gblk(totW), 256, 0, stream>>>(w.aEnc, 51200, w.am2, 256, 1, 0,
                                         FIN(25), w.mA + 30, LLn*40, 40, totW);
  k_wcos<<<gblk(totW), 256, 0, stream>>>(w.bEnc, 51200, w.aEnc + 255*200, 51200, 0, 1,
                                         FIN(22), w.mB + 0,  LLn*40, 40, totW);
  k_wcos<<<gblk(totW), 256, 0, stream>>>(w.bEnc, 51200, w.s1m, 51200, 200, 1,
                                         FIN(24), w.mB + 20, LLn*40, 40, totW);
  k_wcos<<<gblk(totW), 256, 0, stream>>>(w.bEnc, 51200, w.am1, 256, 1, 0,
                                         FIN(25), w.mB + 30, LLn*40, 40, totW);
  // max-pool matching
  k_wn<<<gblk(totW), 256, 0, stream>>>(w.aEnc, FIN(23), w.wnA, totW);
  k_wn<<<gblk(totW), 256, 0, stream>>>(w.bEnc, FIN(23), w.wnB, totW);
  k_fill<<<gblk((long)BLr*10), 256, 0, stream>>>(w.mB + 10, BLr, 40, 10, -1e30f);
  k_mpm<<<dim3(16, PPn, BB), 256, 0, stream>>>(w.aEnc, w.bEnc, FIN(23), w.wnA, w.wnB,
                                               w.mA + 10, w.mB + 10);

  // ---- fusion + pooling (writes directly into z) ----
  fusion_side(stream, d_in, w, w.aEnc, w.fa, w.mA, w.z + 0);
  fusion_side(stream, d_in, w, w.bEnc, w.fb, w.mB, w.z + 800);

  // ---- prediction ----
  launch_gemm(stream, w.z,  FIN(46), FIN(47), w.ph,          64, 400, 1600, 1600, 1600, 400,
              0,0,1, 1.f, 1,0,0,0);
  launch_gemm(stream, w.ph, FIN(48), FIN(49), (float*)d_out, 64, 2,   400,  400,  400,  2,
              0,0,0, 1.f, 1,0,0,0);
  #undef FIN
}
